// GINConvFFT_31044023616144
// MI455X (gfx1250) — compile-verified, matrix operand order checked by reading, not run on hardware
//
#include <hip/hip_runtime.h>

// ---------------- types ----------------
typedef __bf16 bf16_t;
typedef __attribute__((ext_vector_type(16))) __bf16 bf16x16;
typedef __attribute__((ext_vector_type(8)))  float  f32x8;

union FragU { bf16x16 v; uint4 q[2]; };
union Pack8 { uint4 q; bf16_t h[8]; };

// Problem constants
#define Bq   32
#define Nn   512
#define LH   768     // L*H
#define OUTC 64
#define ROWS (Bq*Nn) // 16384

#define LDA 40       // padded LDS row stride (bf16 elems) -> 80B, 16B aligned
#define LDB 40

// Async copy of 16B global -> LDS (per-lane), tracked by ASYNCcnt.
// VDST = VGPR holding LDS byte address (addr[31:0] of the generic pointer),
// VADDR = 64-bit global address.
__device__ __forceinline__ void async_copy_b128(bf16_t* lds_dst, const bf16_t* gsrc) {
    unsigned lds_addr = (unsigned)(uintptr_t)lds_dst;
    asm volatile("global_load_async_to_lds_b128 %0, %1, off"
                 :: "v"(lds_addr), "v"(gsrc) : "memory");
}
__device__ __forceinline__ void async_wait0() {
    asm volatile("s_wait_asynccnt 0x0" ::: "memory");
}

// Load one 16x32 bf16 fragment (A-layout) from row-major memory with leading dim ld.
// Per ISA 7.12.2: lane L holds row m=L%16; elems 0..7 -> K = 8*(L/16)+0..7,
// elems 8..15 -> K = 16 + 8*(L/16) + 0..7  => two contiguous 16B chunks.
__device__ __forceinline__ bf16x16 load_frag16(const bf16_t* base, int ld, int lane) {
    int m  = lane & 15;
    int hi = (lane >> 4) & 1;
    const bf16_t* r = base + m * ld + hi * 8;
    FragU f;
    f.q[0] = *(const uint4*)(r);
    f.q[1] = *(const uint4*)(r + 16);
    return f.v;
}

// ---------------- conversions / repacks ----------------
__global__ void k_cvt_f32_bf16(const float* __restrict__ src, bf16_t* __restrict__ dst, int n) {
    int i = blockIdx.x * blockDim.x + threadIdx.x;
    if (i < n) dst[i] = (bf16_t)src[i];
}

// Wk[k][l*64+h][o] = weight[l][h*4+k][o]; weight is [12][256][64]
__global__ void k_repack_wk(const float* __restrict__ weight, bf16_t* __restrict__ Wk) {
    int i = blockIdx.x * blockDim.x + threadIdx.x;
    if (i >= 4 * LH * OUTC) return;
    int o = i & 63;
    int r = (i / OUTC) % LH;          // l*64+h
    int k = i / (LH * OUTC);          // hop index
    int l = r >> 6;
    int h = r & 63;
    Wk[i] = (bf16_t)weight[(l * 256 + (h * 4 + k)) * OUTC + o];
}

// ---------------- diffusion hop: O = (1+eps)*X + A @ X  ----------------
// A: [512][512] bf16; Xbf/Xf32: [B][512][768].
// Block tile 128x128, 8 waves of 32x64 (2x4 frags -> 8 WMMAs/K-step).
#define DT_BM 128
#define DT_BN 128
#define DT_BK 32
template<bool WRITE_F32>
__global__ __launch_bounds__(256)
void k_diffuse(const bf16_t* __restrict__ A,
               const bf16_t* __restrict__ Xbf,
               const float*  __restrict__ Xf32,
               const float*  __restrict__ epsp,
               float*        __restrict__ Of32,
               bf16_t*       __restrict__ Obf)
{
    __shared__ bf16_t sA[DT_BM * LDA];   // 10240 B
    __shared__ bf16_t sB[DT_BN * LDB];   // 10240 B (transposed X tile: [d][k])
    const int b   = blockIdx.z;
    const int r0  = blockIdx.y * DT_BM;  // n base
    const int c0  = blockIdx.x * DT_BN;  // d base
    const int tid = threadIdx.x;
    const int lane = tid & 31;
    const int wave = tid >> 5;           // 8 waves: 4 (rows of 32) x 2 (cols of 64)
    const int wr = wave >> 1;            // 0..3
    const int wc = wave & 1;             // 0..1

    const float ope = 1.0f + epsp[0];
    const bf16_t* Xb = Xbf + (size_t)b * Nn * LH;

    f32x8 acc[2][4] = {};

    for (int k0 = 0; k0 < Nn; k0 += DT_BK) {
        __syncthreads();
        // A tile 128x32: 512 x b128 async copies global->LDS (no VGPR staging)
        #pragma unroll
        for (int i = 0; i < 2; ++i) {
            int idx = tid + i * 256;
            int r  = idx >> 2;
            int kq = idx & 3;
            async_copy_b128(sA + r * LDA + kq * 8,
                            A + (size_t)(r0 + r) * Nn + k0 + kq * 8);
        }
        // X tile 32x128 -> transposed LDS [d_local][k_local]
        #pragma unroll
        for (int i = 0; i < 2; ++i) {
            int idx = tid + i * 256;       // 0..511
            int k   = idx >> 4;            // 0..31
            int dq  = (idx & 15) * 8;      // 0..120
            Pack8 p;
            p.q = *(const uint4*)(Xb + (size_t)(k0 + k) * LH + c0 + dq);
            #pragma unroll
            for (int j = 0; j < 8; ++j) sB[(dq + j) * LDB + k] = p.h[j];
        }
        // prefetch next K-step tiles toward the WGP (global_prefetch_b8)
        if (k0 + DT_BK < Nn) {
            __builtin_prefetch(A + (size_t)(r0 + (tid >> 1)) * Nn + k0 + DT_BK, 0, 2);
            __builtin_prefetch(Xb + (size_t)(k0 + DT_BK + (tid >> 3)) * LH + c0 + (tid & 7) * 16, 0, 2);
        }
        async_wait0();
        __syncthreads();

        bf16x16 afrag[2], bfrag[4];
        #pragma unroll
        for (int tm = 0; tm < 2; ++tm)
            afrag[tm] = load_frag16(sA + (wr * 32 + tm * 16) * LDA, LDA, lane);
        #pragma unroll
        for (int tn = 0; tn < 4; ++tn)
            bfrag[tn] = load_frag16(sB + (wc * 64 + tn * 16) * LDB, LDB, lane);
        #pragma unroll
        for (int tm = 0; tm < 2; ++tm)
            #pragma unroll
            for (int tn = 0; tn < 4; ++tn)
                acc[tm][tn] = __builtin_amdgcn_wmma_f32_16x16x32_bf16(
                    false, afrag[tm], false, bfrag[tn], (short)0, acc[tm][tn], false, false);
    }

    // epilogue: + (1+eps)*X, write f32 (compile-time) and bf16 (always)
    const float* Xfb = Xf32 + (size_t)b * Nn * LH;
    float*  Ofb = Of32 + (size_t)b * Nn * LH;
    bf16_t* Obb = Obf + (size_t)b * Nn * LH;
    const int nlo = lane & 15;
    const int mhi = (lane >> 4) * 8;
    #pragma unroll
    for (int tm = 0; tm < 2; ++tm)
        #pragma unroll
        for (int tn = 0; tn < 4; ++tn) {
            const int col = c0 + wc * 64 + tn * 16 + nlo;
            const int rowb = r0 + wr * 32 + tm * 16 + mhi;
            // batch the residual loads first, then compute + store
            float xv[8];
            #pragma unroll
            for (int r = 0; r < 8; ++r)
                xv[r] = Xfb[(size_t)(rowb + r) * LH + col];
            #pragma unroll
            for (int r = 0; r < 8; ++r) {
                size_t off = (size_t)(rowb + r) * LH + col;
                float v = acc[tm][tn][r] + ope * xv[r];
                if (WRITE_F32) Ofb[off] = v;
                Obb[off] = (bf16_t)v;
            }
        }
}

// ---------------- mixing GEMM: out1[16384,64] = sum_k hop_k @ Wk ----------------
#define S3_BM 128
__global__ __launch_bounds__(256)
void k_mixw(const bf16_t* __restrict__ hops,  // [4][16384][768]
            const bf16_t* __restrict__ Wk,    // [4][768][64]
            float*        __restrict__ out1)  // [16384][64]
{
    __shared__ bf16_t sA[S3_BM * LDA];   // 10240 B
    __shared__ bf16_t sB[OUTC * LDB];    // 5120 B ([o][k])
    const int r0  = blockIdx.x * S3_BM;
    const int tid = threadIdx.x;
    const int lane = tid & 31;
    const int wave = tid >> 5;    // 8 waves: 4 (rows) x 2 (cols)
    const int wr = wave >> 1;
    const int wc = wave & 1;

    f32x8 acc[2][2] = {};
    for (int h = 0; h < 4; ++h) {
        const bf16_t* Xh = hops + (size_t)h * ROWS * LH;
        const bf16_t* Wh = Wk   + (size_t)h * LH * OUTC;
        for (int k0 = 0; k0 < LH; k0 += 32) {
            __syncthreads();
            // A tile 128x32: async global->LDS copies
            #pragma unroll
            for (int i = 0; i < 2; ++i) {
                int idx = tid + i * 256;
                int r = idx >> 2, kq = idx & 3;
                async_copy_b128(sA + r * LDA + kq * 8,
                                Xh + (size_t)(r0 + r) * LH + k0 + kq * 8);
            }
            // B tile 32x64 -> transposed [o][k]
            {
                int k  = tid >> 3;        // 0..31
                int oq = (tid & 7) * 8;   // 0..56
                Pack8 p;
                p.q = *(const uint4*)(Wh + (size_t)(k0 + k) * OUTC + oq);
                #pragma unroll
                for (int j = 0; j < 8; ++j) sB[(oq + j) * LDB + k] = p.h[j];
            }
            if (k0 + 32 < LH)
                __builtin_prefetch(Xh + (size_t)(r0 + (tid >> 1)) * LH + k0 + 32, 0, 2);
            async_wait0();
            __syncthreads();

            bf16x16 afrag[2], bfrag[2];
            #pragma unroll
            for (int tm = 0; tm < 2; ++tm)
                afrag[tm] = load_frag16(sA + (wr * 32 + tm * 16) * LDA, LDA, lane);
            #pragma unroll
            for (int tn = 0; tn < 2; ++tn)
                bfrag[tn] = load_frag16(sB + (wc * 32 + tn * 16) * LDB, LDB, lane);
            #pragma unroll
            for (int tm = 0; tm < 2; ++tm)
                #pragma unroll
                for (int tn = 0; tn < 2; ++tn)
                    acc[tm][tn] = __builtin_amdgcn_wmma_f32_16x16x32_bf16(
                        false, afrag[tm], false, bfrag[tn], (short)0, acc[tm][tn], false, false);
        }
    }
    const int nlo = lane & 15;
    const int mhi = (lane >> 4) * 8;
    #pragma unroll
    for (int tm = 0; tm < 2; ++tm)
        #pragma unroll
        for (int tn = 0; tn < 2; ++tn) {
            int col = wc * 32 + tn * 16 + nlo;
            #pragma unroll
            for (int r = 0; r < 8; ++r) {
                int row = r0 + wr * 32 + tm * 16 + mhi + r;
                out1[(size_t)row * OUTC + col] = acc[tm][tn][r];
            }
        }
}

// ---------------- BN stats: deterministic two-level reduction ----------------
// partials: [32 blocks][2][64]
__global__ __launch_bounds__(256)
void k_stats_partial(const float* __restrict__ out1, float* __restrict__ partials) {
    __shared__ float ssum[256], ssq[256];
    const int c  = threadIdx.x & 63;
    const int rg = threadIdx.x >> 6;          // 0..3
    const int rbase = blockIdx.x * 512;
    float s = 0.f, q = 0.f;
    for (int r = rg; r < 512; r += 4) {
        float v = out1[(size_t)(rbase + r) * OUTC + c];
        s += v; q += v * v;
    }
    ssum[threadIdx.x] = s; ssq[threadIdx.x] = q;
    __syncthreads();
    if (threadIdx.x < 64) {
        float S = ssum[c] + ssum[c + 64] + ssum[c + 128] + ssum[c + 192];
        float Q = ssq[c]  + ssq[c + 64]  + ssq[c + 128]  + ssq[c + 192];
        partials[(size_t)blockIdx.x * 128 + c]       = S;
        partials[(size_t)blockIdx.x * 128 + 64 + c]  = Q;
    }
}

__global__ void k_stats_reduce(const float* __restrict__ partials, float* __restrict__ stats) {
    int t = threadIdx.x;                      // 0..127
    if (t >= 128) return;
    float s = 0.f;
    for (int j = 0; j < 32; ++j) s += partials[(size_t)j * 128 + t];
    stats[t] = s;                             // [0..63]=sum, [64..127]=sumsq
}

// ---------------- BN + ReLU -> bf16 ----------------
__global__ void k_bnrelu(const float* __restrict__ out1, const float* __restrict__ stats,
                         const float* __restrict__ gamma, const float* __restrict__ beta,
                         bf16_t* __restrict__ y) {
    int i = blockIdx.x * blockDim.x + threadIdx.x;
    if (i >= ROWS * OUTC) return;
    int c = i & 63;
    float mean = stats[c] * (1.0f / (float)ROWS);
    float var  = stats[64 + c] * (1.0f / (float)ROWS) - mean * mean;
    float v = (out1[i] - mean) * rsqrtf(var + 1e-5f) * gamma[c] + beta[c];
    y[i] = (bf16_t)fmaxf(v, 0.0f);
}

// ---------------- final GEMM: out[bn,o] = sum_c y[bn,c]*w2[o,c] + b2[o] ----------
// w2bf is w2 converted to bf16 (layout [o][c]) == exactly the Bt[n=o][k=c] layout needed.
__global__ __launch_bounds__(256)
void k_fc2(const bf16_t* __restrict__ y,     // [16384][64]
           const bf16_t* __restrict__ w2bf,  // [64][64]
           const float*  __restrict__ b2,
           float*        __restrict__ out)   // [16384][64]
{
    const int r0  = blockIdx.x * 128;
    const int tid = threadIdx.x;
    const int lane = tid & 31;
    const int wave = tid >> 5;     // 4 (rows) x 2 (cols)
    const int wr = wave >> 1;
    const int wc = wave & 1;

    f32x8 acc[2][2] = {};
    #pragma unroll
    for (int k0 = 0; k0 < OUTC; k0 += 32) {
        bf16x16 afrag[2], bfrag[2];
        #pragma unroll
        for (int tm = 0; tm < 2; ++tm)
            afrag[tm] = load_frag16(y + (size_t)(r0 + wr * 32 + tm * 16) * OUTC + k0, OUTC, lane);
        #pragma unroll
        for (int tn = 0; tn < 2; ++tn)
            bfrag[tn] = load_frag16(w2bf + (size_t)(wc * 32 + tn * 16) * OUTC + k0, OUTC, lane);
        #pragma unroll
        for (int tm = 0; tm < 2; ++tm)
            #pragma unroll
            for (int tn = 0; tn < 2; ++tn)
                acc[tm][tn] = __builtin_amdgcn_wmma_f32_16x16x32_bf16(
                    false, afrag[tm], false, bfrag[tn], (short)0, acc[tm][tn], false, false);
    }
    const int nlo = lane & 15;
    const int mhi = (lane >> 4) * 8;
    #pragma unroll
    for (int tm = 0; tm < 2; ++tm)
        #pragma unroll
        for (int tn = 0; tn < 2; ++tn) {
            int col = wc * 32 + tn * 16 + nlo;
            float bb = b2[col];
            #pragma unroll
            for (int r = 0; r < 8; ++r) {
                int row = r0 + wr * 32 + tm * 16 + mhi + r;
                out[(size_t)row * OUTC + col] = acc[tm][tn][r] + bb;
            }
        }
}

// ---------------- host-side orchestration ----------------
extern "C" void kernel_launch(void* const* d_in, const int* in_sizes, int n_in,
                              void* d_out, int out_size, void* d_ws, size_t ws_size,
                              hipStream_t stream) {
    const float* x       = (const float*)d_in[0];   // [32,512,12,64] == [32,512,768]
    const float* support = (const float*)d_in[1];   // [2,512,512]
    const float* weight  = (const float*)d_in[2];   // [12,256,64]
    const float* epsp    = (const float*)d_in[3];   // [1]
    const float* gamma   = (const float*)d_in[4];   // [64]
    const float* beta    = (const float*)d_in[5];   // [64]
    const float* w2      = (const float*)d_in[6];   // [64,64]
    const float* b2      = (const float*)d_in[7];   // [64]
    float* out = (float*)d_out;                     // [32,512,64]

    // workspace carve-up
    char* ws = (char*)d_ws;
    size_t off = 0;
    auto carve = [&](size_t bytes) -> void* {
        void* p = ws + off;
        off += (bytes + 255) & ~(size_t)255;
        return p;
    };
    const size_t nX = (size_t)Bq * Nn * LH;         // 12,582,912
    bf16_t* xbf   = (bf16_t*)carve(nX * sizeof(bf16_t));
    bf16_t* Abf   = (bf16_t*)carve((size_t)2 * Nn * Nn * sizeof(bf16_t));
    bf16_t* Wk    = (bf16_t*)carve((size_t)4 * LH * OUTC * sizeof(bf16_t));
    bf16_t* w2bf  = (bf16_t*)carve((size_t)OUTC * OUTC * sizeof(bf16_t));
    float*  tmpf  = (float*) carve(nX * sizeof(float));
    bf16_t* hops  = (bf16_t*)carve((size_t)4 * nX * sizeof(bf16_t));
    float*  out1  = (float*) carve((size_t)ROWS * OUTC * sizeof(float));
    float*  parts = (float*) carve((size_t)32 * 128 * sizeof(float));
    float*  stats = (float*) carve((size_t)128 * sizeof(float));
    bf16_t* ybf   = (bf16_t*)carve((size_t)ROWS * OUTC * sizeof(bf16_t));
    (void)ws_size; (void)in_sizes; (void)n_in; (void)out_size;

    // 1) conversions / repacks
    k_cvt_f32_bf16<<<(int)((nX + 255) / 256), 256, 0, stream>>>(x, xbf, (int)nX);
    k_cvt_f32_bf16<<<(2 * Nn * Nn + 255) / 256, 256, 0, stream>>>(support, Abf, 2 * Nn * Nn);
    k_repack_wk<<<(4 * LH * OUTC + 255) / 256, 256, 0, stream>>>(weight, Wk);
    k_cvt_f32_bf16<<<(OUTC * OUTC + 255) / 256, 256, 0, stream>>>(w2, w2bf, OUTC * OUTC);

    // 2) diffusion hops (WMMA GEMMs)
    dim3 dgrid(LH / DT_BN, Nn / DT_BM, Bq);   // (6, 4, 32)
    const bf16_t* A0 = Abf;
    const bf16_t* A1 = Abf + (size_t)Nn * Nn;
    bf16_t* hop0 = hops;
    bf16_t* hop1 = hops + nX;
    bf16_t* hop2 = hops + 2 * nX;
    bf16_t* hop3 = hops + 3 * nX;
    // support 0: hop0 feeds hop1, so hop0 also writes f32
    k_diffuse<true ><<<dgrid, 256, 0, stream>>>(A0, xbf,  x,    epsp, tmpf, hop0);
    k_diffuse<false><<<dgrid, 256, 0, stream>>>(A0, hop0, tmpf, epsp, tmpf, hop1);
    // support 1
    k_diffuse<true ><<<dgrid, 256, 0, stream>>>(A1, xbf,  x,    epsp, tmpf, hop2);
    k_diffuse<false><<<dgrid, 256, 0, stream>>>(A1, hop2, tmpf, epsp, tmpf, hop3);

    // 3) mixing GEMM
    k_mixw<<<ROWS / S3_BM, 256, 0, stream>>>(hops, Wk, out1);

    // 4) BatchNorm stats (deterministic), normalize + ReLU
    k_stats_partial<<<32, 256, 0, stream>>>(out1, parts);
    k_stats_reduce<<<1, 128, 0, stream>>>(parts, stats);
    k_bnrelu<<<(ROWS * OUTC + 255) / 256, 256, 0, stream>>>(out1, stats, gamma, beta, ybf);

    // 5) final GEMM + bias -> d_out
    k_fc2<<<ROWS / 128, 256, 0, stream>>>(ybf, w2bf, b2, out);
}
